// CausalSelfAttention_14929306321255
// MI455X (gfx1250) — matrix operand order repaired and checked by reading, not compiled, not run on hardware
//
#include <hip/hip_runtime.h>
#include <hip/hip_bf16.h>

typedef __attribute__((ext_vector_type(16))) _Float16 v16h;
typedef __attribute__((ext_vector_type(8)))  _Float16 v8h;
typedef __attribute__((ext_vector_type(8)))  float    v8f;
typedef __attribute__((ext_vector_type(4)))  int      v4i;

union AFrag { v16h v; v8h h[2]; };

#define BATCH 2
#define TSEQ 2048
#define CDIM 1024
#define NHEAD 16
#define HDIM 64
#define D2 128

// ---------------------------------------------------------------------------
// CDNA5 async global->LDS copy (GLOBAL_LOAD_ASYNC_TO_LDS_B128, ASYNCcnt),
// with a synchronous fallback if the builtin is unavailable in this toolchain.
// Probe result: builtin signature takes (int4 AS1*, int4 AS3*, imm, imm).
// ---------------------------------------------------------------------------
#if __has_builtin(__builtin_amdgcn_global_load_async_to_lds_b128) && \
    __has_builtin(__builtin_amdgcn_s_wait_asynccnt)
#define USE_ASYNC_COPY 1
#else
#define USE_ASYNC_COPY 0
#endif

#if USE_ASYNC_COPY
typedef __attribute__((address_space(1))) v4i* gv4i_p;
typedef __attribute__((address_space(3))) v4i* lv4i_p;
#endif

__device__ __forceinline__ void copy16_to_lds(_Float16* lds_dst, const _Float16* gsrc) {
#if USE_ASYNC_COPY
    v4i* g = (v4i*)const_cast<_Float16*>(gsrc);
    v4i* l = (v4i*)lds_dst;
    __builtin_amdgcn_global_load_async_to_lds_b128((gv4i_p)g, (lv4i_p)l, 0, 0);
#else
    *(uint4*)lds_dst = *(const uint4*)gsrc;
#endif
}

__device__ __forceinline__ void async_copy_wait() {
#if USE_ASYNC_COPY
    __builtin_amdgcn_s_wait_asynccnt(0);
#endif
}

// ---------------------------------------------------------------------------
// f32 -> f16 cast
// ---------------------------------------------------------------------------
__global__ void cast_f32_f16(const float* __restrict__ in, _Float16* __restrict__ out, int n) {
    int i = blockIdx.x * 256 + threadIdx.x;
    if (i < n) out[i] = (_Float16)in[i];
}

// ---------------------------------------------------------------------------
// Generic f16 GEMM: C[M,N] = A[M,K] * B[N,K]^T, f32 accumulate/output.
// Block tile 128x128x32, 8 waves (4x2), each wave 32x64 via 2x4 WMMA tiles.
// Double-buffered LDS with async global->LDS staging.
// ---------------------------------------------------------------------------
__global__ __launch_bounds__(256) void gemm_f16_f32(const _Float16* __restrict__ A,
                                                    const _Float16* __restrict__ Bw,
                                                    float* __restrict__ C,
                                                    int M, int N, int K) {
    __shared__ _Float16 As[2][128 * 32];   // 16KB
    __shared__ _Float16 Bs[2][128 * 32];   // 16KB

    const int tid  = threadIdx.x;
    const int wave = tid >> 5;
    const int lane = tid & 31;
    const int l16  = lane & 15;
    const int hi   = (lane >> 4) & 1;
    const int wm   = wave >> 1;   // 0..3
    const int wn   = wave & 1;    // 0..1
    const int bm   = blockIdx.y * 128;
    const int bn   = blockIdx.x * 128;

    v8f acc[2][4];
    {
        v8f z = {};
        for (int i = 0; i < 2; ++i)
            for (int j = 0; j < 4; ++j) acc[i][j] = z;
    }

    auto load_tile = [&](int k0, int buf) {
        for (int p = 0; p < 2; ++p) {
            int idx = p * 256 + tid;
            int row = idx >> 2;
            int c   = idx & 3;
            copy16_to_lds(&As[buf][row * 32 + c * 8],
                          A + (size_t)(bm + row) * K + k0 + c * 8);
            copy16_to_lds(&Bs[buf][row * 32 + c * 8],
                          Bw + (size_t)(bn + row) * K + k0 + c * 8);
        }
    };

    const int nK = K >> 5;
    load_tile(0, 0);

    for (int kt = 0; kt < nK; ++kt) {
        async_copy_wait();
        __syncthreads();
        if (kt + 1 < nK) load_tile((kt + 1) << 5, (kt + 1) & 1);

        const int buf = kt & 1;
        AFrag af[2];
        v16h  bf[4];
        for (int i = 0; i < 2; ++i) {
            const _Float16* ap = &As[buf][(wm * 32 + i * 16 + l16) * 32 + hi * 16];
            af[i].h[0] = *(const v8h*)ap;
            af[i].h[1] = *(const v8h*)(ap + 8);
        }
        for (int j = 0; j < 4; ++j)
            bf[j] = *(const v16h*)&Bs[buf][(wn * 64 + j * 16 + l16) * 32 + hi * 16];

        for (int i = 0; i < 2; ++i)
            for (int j = 0; j < 4; ++j)
                acc[i][j] = __builtin_amdgcn_wmma_f32_16x16x32_f16(
                    false, af[i].v, false, bf[j], (short)0, acc[i][j], false, false);
    }

    for (int i = 0; i < 2; ++i)
        for (int j = 0; j < 4; ++j) {
            int col = bn + wn * 64 + j * 16 + l16;
            for (int r = 0; r < 8; ++r) {
                int row = bm + wm * 32 + i * 16 + r + hi * 8;
                C[(size_t)row * N + col] = acc[i][j][r];
            }
        }
}

// ---------------------------------------------------------------------------
// PoPE epilogue: qkv(f32, B,T,3C) -> q2/k2 (B,H,T,128 f16), vT (B,H,D,T f16)
// ---------------------------------------------------------------------------
__device__ __forceinline__ float softplusf(float x) {
    return (x > 20.0f) ? x : log1pf(__expf(x));
}

__global__ void pope_kernel(const float* __restrict__ qkv, const float* __restrict__ delta,
                            _Float16* __restrict__ q2, _Float16* __restrict__ k2,
                            _Float16* __restrict__ vtp) {
    int n = blockIdx.x * 256 + threadIdx.x;   // B*T*H*D = 2^22 elements
    int d = n & 63;
    int h = (n >> 6) & 15;
    int t = (n >> 10) & 2047;
    int b = n >> 21;

    size_t rowq = ((size_t)(b * TSEQ + t)) * (3 * CDIM);
    int    c    = h * HDIM + d;
    float q = qkv[rowq + c];
    float k = qkv[rowq + CDIM + c];
    float v = qkv[rowq + 2 * CDIM + c];

    float mu_q = softplusf(q);
    float mu_k = softplusf(k);

    // inv_freq = BASE^(-d/D) = exp(-d * ln(10000)/64)
    float invf = __expf(-(float)d * (9.210340371976184f / 64.0f));
    float fr   = (float)t * invf;
    float cp = cosf(fr), sp = sinf(fr);

    float dl = delta[h * HDIM + d];
    dl = fminf(0.0f, fmaxf(dl, -6.283185307179586f));
    float cd = cosf(dl), sd = sinf(dl);

    size_t bhT = ((size_t)(b * NHEAD + h)) * TSEQ + t;
    q2[bhT * D2 + d]        = (_Float16)(mu_q * cp);
    q2[bhT * D2 + HDIM + d] = (_Float16)(mu_q * sp);
    k2[bhT * D2 + d]        = (_Float16)(mu_k * (cp * cd - sp * sd));
    k2[bhT * D2 + HDIM + d] = (_Float16)(mu_k * (sp * cd + cp * sd));
    vtp[(((size_t)(b * NHEAD + h)) * HDIM + d) * TSEQ + t] = (_Float16)v;
}

// ---------------------------------------------------------------------------
// Flash attention. Grid: B*H*(T/64) workgroups of 128 threads (4 waves).
// Each wave owns 16 query rows; K' and V^T tiles of 64 keys double-buffered
// in LDS via async copies.
// ---------------------------------------------------------------------------
__global__ __launch_bounds__(128) void flash_attn_kernel(const _Float16* __restrict__ q2,
                                                         const _Float16* __restrict__ k2,
                                                         const _Float16* __restrict__ vtp,
                                                         _Float16* __restrict__ yh) {
    __shared__ _Float16 Ks[2][64 * D2];    // [s][d2]   32KB
    __shared__ _Float16 Vts[2][HDIM * 64]; // [d][s]    16KB
    __shared__ _Float16 Ps[4 * 16 * 64];   // per-wave P 8KB

    const int tid  = threadIdx.x;
    const int wave = tid >> 5;
    const int lane = tid & 31;
    const int l16  = lane & 15;
    const int hi   = (lane >> 4) & 1;

    const int qblk  = blockIdx.x & (TSEQ / 64 - 1);
    const int bh    = blockIdx.x >> 5;          // T/64 == 32 blocks per (b,h)
    const int h     = bh & (NHEAD - 1);
    const int b     = bh >> 4;
    const int qbase = qblk * 64;
    const int wrow  = qbase + wave * 16;        // first query row of this wave

    const _Float16* qb = q2 + (size_t)bh * TSEQ * D2;
    const _Float16* kb = k2 + (size_t)bh * TSEQ * D2;
    const _Float16* vb = vtp + (size_t)bh * HDIM * TSEQ;

    // Q fragments for all 4 K-steps (K dim = 128) stay in registers.
    AFrag qf[4];
    {
        const _Float16* qp = qb + (size_t)(wrow + l16) * D2;
        for (int ks = 0; ks < 4; ++ks) {
            int k0 = ks * 32 + hi * 16;
            qf[ks].h[0] = *(const v8h*)(qp + k0);
            qf[ks].h[1] = *(const v8h*)(qp + k0 + 8);
        }
    }

    v8f   O[4];
    float m_i[8], l_i[8];
    {
        v8f z = {};
        for (int i = 0; i < 4; ++i) O[i] = z;
        for (int r = 0; r < 8; ++r) { m_i[r] = -1e30f; l_i[r] = 0.0f; }
    }

    auto load_kv = [&](int sb, int buf) {
        // K' tile: 64 x 128 halves = 1024 16B chunks / 128 threads.
        for (int p = 0; p < 8; ++p) {
            int idx = p * 128 + tid;
            int row = idx >> 4, c = idx & 15;
            copy16_to_lds(&Ks[buf][row * D2 + c * 8],
                          kb + (size_t)(sb + row) * D2 + c * 8);
        }
        // V^T tile: 64 x 64 halves = 512 chunks.
        for (int p = 0; p < 4; ++p) {
            int idx = p * 128 + tid;
            int dr = idx >> 3, c = idx & 7;
            copy16_to_lds(&Vts[buf][dr * 64 + c * 8],
                          vb + (size_t)dr * TSEQ + sb + c * 8);
        }
    };

    const int nS = qblk + 1;
    load_kv(0, 0);

    for (int sbi = 0; sbi < nS; ++sbi) {
        const int sb = sbi * 64;
        async_copy_wait();
        __syncthreads();
        if (sbi + 1 < nS) load_kv((sbi + 1) * 64, (sbi + 1) & 1);
        const int buf = sbi & 1;

        // Scores: 16x64 tile, K = 128 -> 4 n-tiles x 4 k-steps of WMMA.
        v8f sc[4];
        for (int nt = 0; nt < 4; ++nt) {
            v8f a = {};
            for (int ks = 0; ks < 4; ++ks) {
                v16h bf = *(const v16h*)&Ks[buf][(nt * 16 + l16) * D2 + ks * 32 + hi * 16];
                a = __builtin_amdgcn_wmma_f32_16x16x32_f16(
                        false, qf[ks].v, false, bf, (short)0, a, false, false);
            }
            sc[nt] = a;
        }

        // Scale and causal mask.
        const bool diag = (sb + 63 >= wrow);
        for (int nt = 0; nt < 4; ++nt) {
            int kj = sb + nt * 16 + l16;
            for (int r = 0; r < 8; ++r) {
                float s = sc[nt][r] * 0.125f;   // 1/sqrt(64)
                if (diag) {
                    int qi = wrow + r + hi * 8;
                    if (kj > qi) s = -1e30f;
                }
                sc[nt][r] = s;
            }
        }

        // Online softmax: row stats live across the 16 lanes of each half.
        for (int r = 0; r < 8; ++r) {
            float mx = sc[0][r];
            for (int nt = 1; nt < 4; ++nt) mx = fmaxf(mx, sc[nt][r]);
            for (int off = 1; off < 16; off <<= 1)
                mx = fmaxf(mx, __shfl_xor(mx, off, 32));
            float mnew = fmaxf(m_i[r], mx);
            float corr = __expf(m_i[r] - mnew);
            m_i[r] = mnew;
            float sum = 0.0f;
            for (int nt = 0; nt < 4; ++nt) {
                float pv = __expf(sc[nt][r] - mnew);
                sc[nt][r] = pv;
                sum += pv;
            }
            for (int off = 1; off < 16; off <<= 1)
                sum += __shfl_xor(sum, off, 32);
            l_i[r] = l_i[r] * corr + sum;
            for (int nt = 0; nt < 4; ++nt) O[nt][r] *= corr;
        }

        // Transpose P (C-layout) -> A-fragment layout through per-wave LDS.
        _Float16* pw = &Ps[wave * 16 * 64];
        for (int nt = 0; nt < 4; ++nt)
            for (int r = 0; r < 8; ++r)
                pw[(r + hi * 8) * 64 + nt * 16 + l16] = (_Float16)sc[nt][r];
        __syncthreads();

        // O += P @ V   (K = 64 keys -> 2 k-steps; N = 64 dims -> 4 n-tiles)
        for (int ks2 = 0; ks2 < 2; ++ks2) {
            AFrag pf;
            const _Float16* pp = pw + l16 * 64 + ks2 * 32 + hi * 16;
            pf.h[0] = *(const v8h*)pp;
            pf.h[1] = *(const v8h*)(pp + 8);
            for (int nt2 = 0; nt2 < 4; ++nt2) {
                v16h bf = *(const v16h*)&Vts[buf][(nt2 * 16 + l16) * 64 + ks2 * 32 + hi * 16];
                O[nt2] = __builtin_amdgcn_wmma_f32_16x16x32_f16(
                             false, pf.v, false, bf, (short)0, O[nt2], false, false);
            }
        }
    }

    // Epilogue: y[b, t, h*64 + d] = O / l  (f16 for the output projection GEMM)
    for (int nt2 = 0; nt2 < 4; ++nt2) {
        int col = h * HDIM + nt2 * 16 + l16;
        for (int r = 0; r < 8; ++r) {
            int trow = wrow + r + hi * 8;
            yh[((size_t)(b * TSEQ + trow)) * CDIM + col] = (_Float16)(O[nt2][r] / l_i[r]);
        }
    }
}

// ---------------------------------------------------------------------------
// Host launcher
// ---------------------------------------------------------------------------
extern "C" void kernel_launch(void* const* d_in, const int* in_sizes, int n_in,
                              void* d_out, int out_size, void* d_ws, size_t ws_size,
                              hipStream_t stream) {
    const float* x      = (const float*)d_in[0];   // (2,2048,1024)
    const float* w_attn = (const float*)d_in[1];   // (3072,1024)
    const float* w_proj = (const float*)d_in[2];   // (1024,1024)
    const float* delta  = (const float*)d_in[3];   // (16,64)
    float*       out    = (float*)d_out;           // (2,2048,1024) f32

    const int BT  = BATCH * TSEQ;                  // 4096
    const int nX  = BT * CDIM;                     // 4,194,304
    const int nWA = 3 * CDIM * CDIM;               // 3,145,728
    const int nWP = CDIM * CDIM;                   // 1,048,576

    char*  ws  = (char*)d_ws;
    size_t off = 0;
    auto carve = [&](size_t bytes) -> char* {
        char* p = ws + off;
        off += (bytes + 255) & ~(size_t)255;
        return p;
    };
    _Float16* x_h     = (_Float16*)carve((size_t)nX * 2);
    _Float16* wattn_h = (_Float16*)carve((size_t)nWA * 2);
    _Float16* wproj_h = (_Float16*)carve((size_t)nWP * 2);
    float*    qkv     = (float*)   carve((size_t)BT * 3 * CDIM * 4);
    _Float16* q2      = (_Float16*)carve((size_t)BATCH * NHEAD * TSEQ * D2 * 2);
    _Float16* k2      = (_Float16*)carve((size_t)BATCH * NHEAD * TSEQ * D2 * 2);
    _Float16* vtp     = (_Float16*)carve((size_t)BATCH * NHEAD * HDIM * TSEQ * 2);
    _Float16* y_h     = (_Float16*)carve((size_t)nX * 2);

    cast_f32_f16<<<(nX + 255) / 256, 256, 0, stream>>>(x, x_h, nX);
    cast_f32_f16<<<(nWA + 255) / 256, 256, 0, stream>>>(w_attn, wattn_h, nWA);
    cast_f32_f16<<<(nWP + 255) / 256, 256, 0, stream>>>(w_proj, wproj_h, nWP);

    // qkv = x @ w_attn^T : (4096 x 3072), K = 1024
    gemm_f16_f32<<<dim3(3 * CDIM / 128, BT / 128), 256, 0, stream>>>(
        x_h, wattn_h, qkv, BT, 3 * CDIM, CDIM);

    pope_kernel<<<nX / 256, 256, 0, stream>>>(qkv, delta, q2, k2, vtp);

    flash_attn_kernel<<<BATCH * NHEAD * (TSEQ / 64), 128, 0, stream>>>(q2, k2, vtp, y_h);

    // out = y @ w_proj^T : (4096 x 1024), K = 1024, f32 output
    gemm_f16_f32<<<dim3(CDIM / 128, BT / 128), 256, 0, stream>>>(
        y_h, wproj_h, out, BT, CDIM, CDIM);
}
